// MultiHeadSelfAttention_86706799772148
// MI455X (gfx1250) — matrix operand order repaired and checked by reading, not compiled, not run on hardware
//
#include <hip/hip_runtime.h>
#include <hip/hip_bf16.h>
#include <stdint.h>

// Problem dims (fixed by reference)
#define DM   1024
#define NH   16
#define DK   64
#define B_   2
#define S_   2048
#define MTOT (B_*S_)   // 4096

typedef __attribute__((ext_vector_type(16))) __bf16 v16bf;
typedef __attribute__((ext_vector_type(8)))  float  v8f;
typedef __attribute__((ext_vector_type(4)))  unsigned int u32x4;
typedef __attribute__((ext_vector_type(8)))  int  i32x8;
typedef __attribute__((ext_vector_type(4)))  int  i32x4;
typedef unsigned short u16;
typedef unsigned int   u32;

union FragU { uint4 q[2]; v16bf v; };

__device__ __forceinline__ u16 f2bf(float f) {
  u32 u = __float_as_uint(f);
  u += 0x7FFFu + ((u >> 16) & 1u);   // round-to-nearest-even
  return (u16)(u >> 16);
}

__device__ __forceinline__ v8f vzero() { v8f z = {}; return z; }

// A-fragment (16x32 bf16, row-major source, row stride ld halves):
// lanes 0-15: row m, K halves {0..7}+{16..23}; lanes 16-31: {8..15}+{24..31}
__device__ __forceinline__ v16bf load_a(const u16* base, int row, int ld, int kbase, int lane) {
  const int half = lane >> 4;
  const u16* p = base + (size_t)row * ld + kbase + half * 8;
  FragU u;
  u.q[0] = *(const uint4*)p;
  u.q[1] = *(const uint4*)(p + 16);
  return u.v;
}

// B-fragment (32x16 bf16); column n is a contiguous row in memory (B^T row):
// lanes 0-15: col n, K {0..7}+{16..23}; lanes 16-31: col n, K {8..15}+{24..31}
__device__ __forceinline__ v16bf load_b(const u16* base, int col, int ld, int kbase, int lane) {
  const int half = lane >> 4;
  const u16* p = base + (size_t)col * ld + kbase + half * 8;
  FragU u;
  u.q[0] = *(const uint4*)p;
  u.q[1] = *(const uint4*)(p + 16);
  return u.v;
}

__device__ __forceinline__ v8f wmma_bf16(v16bf a, v16bf b, v8f c) {
  return __builtin_amdgcn_wmma_f32_16x16x32_bf16(false, a, false, b, (short)0, c, false, false);
}

// ---- Tensor Data Mover: 2D tile (bf16 elements) global -> LDS ----
// D# per CDNA5 ISA 8.3/8.4: group0 = {count/flags, lds_addr, global_addr, type=2},
// group1 = {data_size=2B, tensor_dim0/1, tile_dim0/1, tensor_dim0_stride}, groups 2/3 zero (2D).
// This toolchain exposes the 6-arg builtin: (g0, g1, g2, g3, extra8, cpol).
__device__ __forceinline__ void tdm_load_2d_bf16(const u16* gptr, unsigned lds_off,
                                                 unsigned tensor_d0, unsigned tensor_d1,
                                                 unsigned long long d0_stride,
                                                 unsigned tile_d0, unsigned tile_d1) {
  unsigned long long ga = (unsigned long long)(size_t)gptr;
  u32x4 g0;
  g0[0] = 1u;                                            // count=1, user descriptor
  g0[1] = lds_off;                                       // lds_addr (bytes)
  g0[2] = (unsigned)(ga & 0xFFFFFFFFu);                  // global_addr[31:0]
  g0[3] = (unsigned)((ga >> 32) & 0x01FFFFFFu) | (2u << 30);  // addr[56:32] | type=2
  i32x8 g1;
  g1[0] = (int)(1u << 16);                               // data_size=1 -> 2 bytes
  g1[1] = (int)((tensor_d0 & 0xFFFFu) << 16);            // tensor_dim0[15:0]
  g1[2] = (int)(((tensor_d0 >> 16) & 0xFFFFu) | ((tensor_d1 & 0xFFFFu) << 16));
  g1[3] = (int)(((tensor_d1 >> 16) & 0xFFFFu) | ((tile_d0 & 0xFFFFu) << 16));
  g1[4] = (int)(tile_d1 & 0xFFFFu);                      // tile_dim1 (tile_dim2=0)
  g1[5] = (int)(d0_stride & 0xFFFFFFFFull);              // dim0_stride[31:0]
  g1[6] = (int)((d0_stride >> 32) & 0xFFFFull);          // dim0_stride[47:32]
  g1[7] = 0;
  i32x4 gz = {};
  i32x8 gz8 = {};
  __builtin_amdgcn_tensor_load_to_lds(g0, g1, gz, gz, gz8, 0);
}

// ---------------- fp32 -> bf16 convert ----------------
__global__ void cvt_f32_bf16(const float* __restrict__ in, u16* __restrict__ out, int n) {
  int i = blockIdx.x * blockDim.x + threadIdx.x;
  if (i < n) out[i] = f2bf(in[i]);
}

// ---------------- Q/K/V projection GEMM ----------------
// out[m,n] = sum_k xb[m,k]*wb[n,k].  mode 0: Q (scaled 1/sqrt(Dk), [B,H,S,Dk])
// mode 1: K ([B,H,S,Dk])   mode 2: V transposed ([B,H,Dk,S])
__global__ __launch_bounds__(256) void qkv_proj(const u16* __restrict__ xb,
                                                const u16* __restrict__ wb,
                                                u16* __restrict__ out, int mode) {
  const int lane = threadIdx.x & 31;
  const int wave = threadIdx.x >> 5;
  const int l15  = lane & 15;
  const int m0   = blockIdx.y * 128 + wave * 16;
  const int n0   = blockIdx.x * 64;

  v8f acc[4];
#pragma unroll
  for (int t = 0; t < 4; ++t) acc[t] = vzero();

  for (int k = 0; k < DM; k += 32) {
    v16bf a = load_a(xb, m0 + l15, DM, k, lane);
#pragma unroll
    for (int t = 0; t < 4; ++t) {
      v16bf b = load_b(wb, n0 + t * 16 + l15, DM, k, lane);
      acc[t] = wmma_bf16(a, b, acc[t]);
    }
  }

  const int half8 = (lane >> 4) * 8;
  const float scale = (mode == 0) ? 0.125f : 1.0f;  // 1/sqrt(64)
#pragma unroll
  for (int t = 0; t < 4; ++t) {
    int n = n0 + t * 16 + l15;
    int h = n >> 6, d = n & 63;
#pragma unroll
    for (int r = 0; r < 8; ++r) {
      int m = m0 + r + half8;
      int bb = m >> 11, s = m & 2047;
      u16 val = f2bf(acc[t][r] * scale);
      if (mode == 2) out[((size_t)(bb * NH + h) * DK + d) * S_ + s] = val;   // Vt[b,h,d,s]
      else           out[((size_t)(bb * NH + h) * S_ + s) * DK + d] = val;   // [b,h,s,d]
    }
  }
}

// ---------------- flash attention with TDM-staged K/V tiles ----------------
// Block = 8 waves, one (b,h), 128 queries. Per 32-key step, wave 0 DMAs the
// shared K tile (32x64) and V tile (64x32) into LDS via the Tensor Data Mover;
// all waves build WMMA fragments from LDS.
__global__ __launch_bounds__(256) void attn(const u16* __restrict__ Q,
                                            const u16* __restrict__ K,
                                            const u16* __restrict__ Vt,
                                            u16* __restrict__ ctx) {
  __shared__ __align__(16) u16 lk[32][64];     // K tile: keys x d      (4 KB)
  __shared__ __align__(16) u16 lv[64][32];     // V tile: d x keys      (4 KB)
  __shared__ __align__(16) u16 lp[8][16][32];  // per-wave P tiles      (8 KB)

  const int lane  = threadIdx.x & 31;
  const int wave  = threadIdx.x >> 5;
  const int l15   = lane & 15;
  const int half8 = (lane >> 4) * 8;
  const int bh    = blockIdx.y;
  const int q0    = blockIdx.x * 128 + wave * 16;

  const u16* Qb = Q  + (size_t)bh * S_ * DK;
  const u16* Kb = K  + (size_t)bh * S_ * DK;
  const u16* Vb = Vt + (size_t)bh * DK * S_;

  const unsigned lk_off = (unsigned)(size_t)&lk[0][0];
  const unsigned lv_off = (unsigned)(size_t)&lv[0][0];

  v16bf qa0 = load_a(Qb, q0 + l15, DK, 0,  lane);
  v16bf qa1 = load_a(Qb, q0 + l15, DK, 32, lane);

  v8f acc[4];
#pragma unroll
  for (int t = 0; t < 4; ++t) acc[t] = vzero();
  float mr[8], lr[8];
#pragma unroll
  for (int r = 0; r < 8; ++r) { mr[r] = -1e30f; lr[r] = 0.0f; }

  for (int kb = 0; kb < S_; kb += 32) {
    __syncthreads();           // previous iteration's LDS K/V reads complete
    if (wave == 0) {
      // K tile: 32 rows (keys kb..kb+31) x 64 cols, row stride 64 elements
      tdm_load_2d_bf16(Kb + (size_t)kb * DK, lk_off, DK, S_, DK, DK, 32);
      // V tile: 64 rows (d) x 32 cols (keys kb..), row stride 2048 elements
      tdm_load_2d_bf16(Vb + kb, lv_off, S_, DK, S_, 32, DK);
      __builtin_amdgcn_s_wait_tensorcnt(0);
    }
    __syncthreads();           // tiles visible to all waves

    // scores (16q x 32k), contraction over Dk=64 (two k-steps), K from LDS
    v16bf k00 = load_b(&lk[0][0], l15,      DK, 0,  lane);
    v16bf k01 = load_b(&lk[0][0], l15,      DK, 32, lane);
    v16bf k10 = load_b(&lk[0][0], 16 + l15, DK, 0,  lane);
    v16bf k11 = load_b(&lk[0][0], 16 + l15, DK, 32, lane);
    v8f s0 = vzero(); s0 = wmma_bf16(qa0, k00, s0); s0 = wmma_bf16(qa1, k01, s0);
    v8f s1 = vzero(); s1 = wmma_bf16(qa0, k10, s1); s1 = wmma_bf16(qa1, k11, s1);

    // online softmax; C/D layout: VGPR r -> query r (lanes 0-15) / r+8 (lanes 16-31)
#pragma unroll
    for (int r = 0; r < 8; ++r) {
      float t = fmaxf(s0[r], s1[r]);
      t = fmaxf(t, __shfl_xor(t, 1, 32));
      t = fmaxf(t, __shfl_xor(t, 2, 32));
      t = fmaxf(t, __shfl_xor(t, 4, 32));
      t = fmaxf(t, __shfl_xor(t, 8, 32));
      float mn   = fmaxf(mr[r], t);
      float corr = __expf(mr[r] - mn);
      mr[r] = mn;
      float p0 = __expf(s0[r] - mn);
      float p1 = __expf(s1[r] - mn);
      float rs = p0 + p1;
      rs += __shfl_xor(rs, 1, 32);
      rs += __shfl_xor(rs, 2, 32);
      rs += __shfl_xor(rs, 4, 32);
      rs += __shfl_xor(rs, 8, 32);
      lr[r] = lr[r] * corr + rs;
      acc[0][r] *= corr; acc[1][r] *= corr; acc[2][r] *= corr; acc[3][r] *= corr;
      lp[wave][r + half8][l15]      = f2bf(p0);
      lp[wave][r + half8][l15 + 16] = f2bf(p1);
    }
    __syncthreads();           // order P stores before transposed reload

    // reload P in A-fragment layout (16x32), then P @ V with V from LDS
    v16bf pa = load_a(&lp[wave][0][0], l15, 32, 0, lane);
#pragma unroll
    for (int t = 0; t < 4; ++t) {
      v16bf vf = load_b(&lv[0][0], t * 16 + l15, 32, 0, lane);
      acc[t] = wmma_bf16(pa, vf, acc[t]);
    }
  }

  const int bb = bh >> 4, h = bh & 15;
#pragma unroll
  for (int t = 0; t < 4; ++t) {
#pragma unroll
    for (int r = 0; r < 8; ++r) {
      int s = q0 + r + half8;
      float o = acc[t][r] / lr[r];
      ctx[((size_t)(bb * S_ + s)) * DM + h * DK + t * 16 + l15] = f2bf(o);
    }
  }
}

// ---------------- output projection + bias + residual ----------------
__global__ __launch_bounds__(256) void out_proj(const u16* __restrict__ cb,
                                                const u16* __restrict__ wb,
                                                const float* __restrict__ bias,
                                                const float* __restrict__ x,
                                                float* __restrict__ y) {
  const int lane = threadIdx.x & 31;
  const int wave = threadIdx.x >> 5;
  const int l15  = lane & 15;
  const int m0   = blockIdx.y * 128 + wave * 16;
  const int n0   = blockIdx.x * 64;

  v8f acc[4];
#pragma unroll
  for (int t = 0; t < 4; ++t) acc[t] = vzero();

  for (int k = 0; k < DM; k += 32) {
    v16bf a = load_a(cb, m0 + l15, DM, k, lane);
#pragma unroll
    for (int t = 0; t < 4; ++t) {
      v16bf b = load_b(wb, n0 + t * 16 + l15, DM, k, lane);
      acc[t] = wmma_bf16(a, b, acc[t]);
    }
  }

  const int half8 = (lane >> 4) * 8;
#pragma unroll
  for (int t = 0; t < 4; ++t) {
    int n = n0 + t * 16 + l15;
#pragma unroll
    for (int r = 0; r < 8; ++r) {
      int m = m0 + r + half8;
      y[(size_t)m * DM + n] = acc[t][r] + bias[n] + x[(size_t)m * DM + n];
    }
  }
}

// ---------------- LayerNorm per row ----------------
__global__ __launch_bounds__(256) void layernorm(const float* __restrict__ y,
                                                 const float* __restrict__ g,
                                                 const float* __restrict__ b,
                                                 float* __restrict__ out) {
  __shared__ float s1[256], s2[256];
  const int row = blockIdx.x;
  const float* yr = y + (size_t)row * DM;
  float sum = 0.f, sq = 0.f;
  for (int c = threadIdx.x; c < DM; c += 256) {
    float v = yr[c];
    sum += v; sq += v * v;
  }
  s1[threadIdx.x] = sum; s2[threadIdx.x] = sq;
  __syncthreads();
  for (int off = 128; off > 0; off >>= 1) {
    if (threadIdx.x < off) {
      s1[threadIdx.x] += s1[threadIdx.x + off];
      s2[threadIdx.x] += s2[threadIdx.x + off];
    }
    __syncthreads();
  }
  float mean = s1[0] * (1.0f / DM);
  float var  = s2[0] * (1.0f / DM) - mean * mean;
  float inv  = rsqrtf(var + 1e-5f);
  float* outr = out + (size_t)row * DM;
  for (int c = threadIdx.x; c < DM; c += 256) {
    outr[c] = (yr[c] - mean) * inv * g[c] + b[c];
  }
}

extern "C" void kernel_launch(void* const* d_in, const int* in_sizes, int n_in,
                              void* d_out, int out_size, void* d_ws, size_t ws_size,
                              hipStream_t stream) {
  (void)in_sizes; (void)n_in; (void)out_size; (void)ws_size;
  const float* x  = (const float*)d_in[0];
  const float* wq = (const float*)d_in[1];
  const float* wk = (const float*)d_in[2];
  const float* wv = (const float*)d_in[3];
  const float* wo = (const float*)d_in[4];
  const float* bo = (const float*)d_in[5];
  const float* lg = (const float*)d_in[6];
  const float* lb = (const float*)d_in[7];

  char* ws = (char*)d_ws;
  const size_t MB = 1024 * 1024;
  u16* xb  = (u16*)(ws + 0 * MB);   // 8 MB  [4096,1024] bf16
  u16* wqb = (u16*)(ws + 8 * MB);   // 2 MB
  u16* wkb = (u16*)(ws + 10 * MB);  // 2 MB
  u16* wvb = (u16*)(ws + 12 * MB);  // 2 MB
  u16* wob = (u16*)(ws + 14 * MB);  // 2 MB
  u16* Qb  = (u16*)(ws + 16 * MB);  // 8 MB  [B,H,S,Dk] (pre-scaled)
  u16* Kb  = (u16*)(ws + 24 * MB);  // 8 MB  [B,H,S,Dk]
  u16* Vtb = (u16*)(ws + 32 * MB);  // 8 MB  [B,H,Dk,S]
  u16* cb  = (u16*)(ws + 40 * MB);  // 8 MB  ctx [4096,1024] bf16
  float* y = (float*)(ws + 48 * MB); // 16 MB y [4096,1024] fp32

  const int NX = MTOT * DM;   // 4194304
  const int NW = DM * DM;     // 1048576
  cvt_f32_bf16<<<(NX + 255) / 256, 256, 0, stream>>>(x,  xb,  NX);
  cvt_f32_bf16<<<(NW + 255) / 256, 256, 0, stream>>>(wq, wqb, NW);
  cvt_f32_bf16<<<(NW + 255) / 256, 256, 0, stream>>>(wk, wkb, NW);
  cvt_f32_bf16<<<(NW + 255) / 256, 256, 0, stream>>>(wv, wvb, NW);
  cvt_f32_bf16<<<(NW + 255) / 256, 256, 0, stream>>>(wo, wob, NW);

  dim3 gg(DM / 64, MTOT / 128);   // (16, 32)
  qkv_proj<<<gg, 256, 0, stream>>>(xb, wqb, Qb, 0);
  qkv_proj<<<gg, 256, 0, stream>>>(xb, wkb, Kb, 1);
  qkv_proj<<<gg, 256, 0, stream>>>(xb, wvb, Vtb, 2);

  attn<<<dim3(S_ / 128, B_ * NH), 256, 0, stream>>>(Qb, Kb, Vtb, cb);

  out_proj<<<gg, 256, 0, stream>>>(cb, wob, bo, x, y);
  layernorm<<<MTOT, 256, 0, stream>>>(y, lg, lb, (float*)d_out);
}
